// Attention_29884382446443
// MI455X (gfx1250) — matrix operand order repaired and checked by reading, not compiled, not run on hardware
//
#include <hip/hip_runtime.h>
#include <hip/hip_bf16.h>
#include <math.h>

// ---------------------------------------------------------------------------
// Transformer block for MI455X (gfx1250, wave32):
//   - bf16 WMMA 16x16x32 (f32 accum) for all matmuls
//   - Tensor Data Mover (tensor_load_to_lds) stages weight tiles in LDS,
//     double-buffered, tracked with TENSORcnt
//   - flash attention with online softmax, V pre-transposed
// ---------------------------------------------------------------------------

typedef __attribute__((ext_vector_type(16))) __bf16 bf16x16;
typedef __attribute__((ext_vector_type(8)))  __bf16 bf16x8;
typedef __attribute__((ext_vector_type(8)))  float  floatx8;
typedef __attribute__((ext_vector_type(4)))  unsigned int uintx4;
typedef __attribute__((ext_vector_type(8)))  int    intx8;
typedef __attribute__((ext_vector_type(4)))  int    intx4;

#define BATCH 8
#define SEQ   2048
#define DIM   896
#define HEADS 7
#define HDIM  128
#define MTOK  (BATCH * SEQ)   // 16384
#define C3    (3 * DIM)       // 2688
#define FFN   (2 * DIM)       // 1792

static __device__ __forceinline__ floatx8 fzero() {
    floatx8 z;
#pragma unroll
    for (int i = 0; i < 8; ++i) z[i] = 0.0f;
    return z;
}

static __device__ __forceinline__ floatx8 wmma_bf16(bf16x16 a, bf16x16 b, floatx8 c) {
    // D = A(16x32 bf16) * B(32x16 bf16) + C(16x16 f32)
    return __builtin_amdgcn_wmma_f32_16x16x32_bf16(false, a, false, b, (short)0, c,
                                                   false, false);
}

// A-fragment (16x32, row-major source): lane L holds row (L&15);
// element e -> K = klo+e, klo = 16*(L>>4) (matches B fragment order).
static __device__ __forceinline__ bf16x16
load_a_frag(const __bf16* __restrict__ A, int lda, int row0, int k0, int lane) {
    int r   = lane & 15;
    int klo = (lane >> 4) * 16;
    const __bf16* p = A + (size_t)(row0 + r) * lda + k0;
    bf16x8 lo = *(const bf16x8*)(p + klo);
    bf16x8 hi = *(const bf16x8*)(p + klo + 8);
    bf16x16 f;
#pragma unroll
    for (int i = 0; i < 8; ++i) { f[i] = lo[i]; f[i + 8] = hi[i]; }
    return f;
}

// B-fragment (32x16): source stored as B^T row-major ([N][K], e.g. torch
// Linear weight [out,in]). Lane L holds output-column n0+(L&15); element e ->
// K = (L>>4)*16 + e. That is 16 contiguous bf16 per lane.
static __device__ __forceinline__ bf16x16
load_b_frag(const __bf16* __restrict__ Bt, int ldb, int n0, int k0, int lane) {
    int n    = lane & 15;
    int koff = (lane >> 4) * 16;
    const __bf16* p = Bt + (size_t)(n0 + n) * ldb + k0 + koff;
    return *(const bf16x16*)p;
}

// B-fragment from an LDS-staged [64][32] bf16 tile (rows = output columns).
static __device__ __forceinline__ bf16x16
load_b_frag_lds(const __bf16* bbase, int n0, int lane) {
    int n    = n0 + (lane & 15);
    int koff = (lane >> 4) * 16;
    return *(const bf16x16*)(bbase + n * 32 + koff);
}

// ---------------------------------------------------------------------------
// Tensor Data Mover: DMA one 64(rows) x 32(k) bf16 weight tile into LDS.
// D# per CDNA5 ISA ch.8: group0 = {count/flags, lds_addr, global_addr, type=2},
// group1 = {data_size=2B, tensor_dim0=K, tensor_dim1=Nrows, tile=32x64,
// tensor_dim0_stride=K}. Groups 2/3 zero (2D tensor). Tracked by TENSORcnt.
// This toolchain exposes the 6-arg builtin (extra trailing SGPR group).
// ---------------------------------------------------------------------------
static __device__ __forceinline__ void
tdm_load_b_tile(const __bf16* W, int K, int Nrows, int row0, int k0, __bf16* ldsdst) {
    unsigned long long ga = (unsigned long long)(uintptr_t)(W + (size_t)row0 * K + k0);
    unsigned int lds_off  = (unsigned int)(uintptr_t)ldsdst;   // low 32 = LDS byte addr
    uintx4 g0;
    g0[0] = 1u;                                   // count=1 (valid), user mode
    g0[1] = lds_off;                              // lds_addr
    g0[2] = (unsigned int)ga;                     // global_addr[31:0]
    g0[3] = (unsigned int)((ga >> 32) & 0x01FFFFFFu) | (2u << 30);  // addr[56:32]|type=2
    intx8 g1;
    g1[0] = (int)(1u << 16);                      // workgroup_mask=0, data_size=1 (2B)
    g1[1] = (int)(((unsigned)K & 0xFFFFu) << 16); // tensor_dim0[15:0]
    g1[2] = (int)((((unsigned)K >> 16) & 0xFFFFu) |
                  (((unsigned)Nrows & 0xFFFFu) << 16));          // dim0 hi | dim1 lo
    g1[3] = (int)((((unsigned)Nrows >> 16) & 0xFFFFu) | (32u << 16)); // dim1 hi | tile_dim0=32
    g1[4] = 64;                                   // tile_dim1=64 rows, tile_dim2=0
    g1[5] = (int)(unsigned)K;                     // tensor_dim0_stride[31:0] (elements)
    g1[6] = 0;                                    // stride hi | dim1_stride lo
    g1[7] = 0;
    intx4 g2 = {0, 0, 0, 0};
    intx4 g3 = {0, 0, 0, 0};
    intx8 g4 = {0, 0, 0, 0, 0, 0, 0, 0};
    __builtin_amdgcn_tensor_load_to_lds(g0, g1, g2, g3, g4, 0);
}

// ---------------------------------------------------------------------------
// fp32 -> bf16 cast (n must be a multiple of 256)
// ---------------------------------------------------------------------------
__global__ __launch_bounds__(256) void cast_f32_bf16(const float* __restrict__ in,
                                                     __bf16* __restrict__ out, int n) {
    int i = blockIdx.x * 256 + threadIdx.x;
    if (i < n) out[i] = (__bf16)in[i];
}

// ---------------------------------------------------------------------------
// QKV GEMM: [16384 x 896] * [2688 x 896]^T -> scatter into q/k (bh,n,d) and
// v transposed (bh,d,n). Block = 8 waves as 4(M)x2(N) of 32x32 wave tiles.
// Weight tiles staged in LDS by the TDM, double-buffered.
// ---------------------------------------------------------------------------
__global__ __launch_bounds__(256) void gemm_qkv(const __bf16* __restrict__ A,
                                                const __bf16* __restrict__ W,
                                                __bf16* __restrict__ qo,
                                                __bf16* __restrict__ ko,
                                                __bf16* __restrict__ vt) {
    __shared__ __align__(64) __bf16 Blds[2][64][32];   // 8 KB double buffer

    int lane = threadIdx.x & 31, w = threadIdx.x >> 5;
    int wm = w >> 1, wn = w & 1;
    int m0 = blockIdx.x * 128 + wm * 32;
    int nblk = blockIdx.y * 64;
    int n0 = nblk + wn * 32;

    floatx8 acc[2][2];
#pragma unroll
    for (int i = 0; i < 2; ++i)
#pragma unroll
        for (int j = 0; j < 2; ++j) acc[i][j] = fzero();

    const int ksteps = DIM / 32;   // 28
    if (w == 0) tdm_load_b_tile(W, DIM, C3, nblk, 0, &Blds[0][0][0]);

    auto compute = [&](int i) {
        int k0 = i * 32;
        const __bf16* bbase = &Blds[i & 1][0][0];
        bf16x16 a0 = load_a_frag(A, DIM, m0, k0, lane);
        bf16x16 a1 = load_a_frag(A, DIM, m0 + 16, k0, lane);
        bf16x16 b0 = load_b_frag_lds(bbase, wn * 32, lane);
        bf16x16 b1 = load_b_frag_lds(bbase, wn * 32 + 16, lane);
        acc[0][0] = wmma_bf16(a0, b0, acc[0][0]);
        acc[0][1] = wmma_bf16(a0, b1, acc[0][1]);
        acc[1][0] = wmma_bf16(a1, b0, acc[1][0]);
        acc[1][1] = wmma_bf16(a1, b1, acc[1][1]);
    };

    for (int i = 0; i < ksteps - 1; ++i) {
        if (w == 0) {
            tdm_load_b_tile(W, DIM, C3, nblk, (i + 1) * 32, &Blds[(i + 1) & 1][0][0]);
            __builtin_amdgcn_s_wait_tensorcnt(1);   // tile i complete (in-order)
        }
        __syncthreads();
        compute(i);
        __syncthreads();
    }
    if (w == 0) __builtin_amdgcn_s_wait_tensorcnt(0);
    __syncthreads();
    compute(ksteps - 1);

    int half = lane >> 4, cn = lane & 15;
#pragma unroll
    for (int i = 0; i < 2; ++i) {
#pragma unroll
        for (int j = 0; j < 2; ++j) {
            int col = n0 + j * 16 + cn;          // 0..2687
            int s   = col / DIM;                 // 0=q 1=k 2=v
            int rem = col - s * DIM;
            int hh  = rem >> 7;                  // head
            int dd  = rem & 127;                 // within-head dim
#pragma unroll
            for (int v = 0; v < 8; ++v) {
                int row  = m0 + i * 16 + v + 8 * half;   // token index
                int bidx = row >> 11;
                int nn   = row & 2047;
                size_t bh = (size_t)(bidx * HEADS + hh);
                __bf16 val = (__bf16)acc[i][j][v];
                if (s == 0)
                    qo[(bh * SEQ + nn) * HDIM + dd] = val;
                else if (s == 1)
                    ko[(bh * SEQ + nn) * HDIM + dd] = val;
                else
                    vt[(bh * HDIM + dd) * SEQ + nn] = val;   // V transposed
            }
        }
    }
}

// ---------------------------------------------------------------------------
// Flash attention. Grid: (SEQ/128, BATCH*HEADS), 8 waves; each wave owns 16
// query rows and the full d=128 output (8 f32 C-tiles). Keys streamed in
// chunks of 32; P transposed C->A layout through per-wave LDS.
// ---------------------------------------------------------------------------
__global__ __launch_bounds__(256) void attn_flash(const __bf16* __restrict__ q,
                                                  const __bf16* __restrict__ k,
                                                  const __bf16* __restrict__ vt,
                                                  float* __restrict__ attn) {
    __shared__ __align__(64) __bf16 Plds[8][16][32];   // per-wave P staging

    int lane = threadIdx.x & 31, w = threadIdx.x >> 5;
    int bh = blockIdx.y;
    int bidx = bh / HEADS, hh = bh - bidx * HEADS;

    const __bf16* qp = q  + (size_t)bh * SEQ * HDIM;
    const __bf16* kp = k  + (size_t)bh * SEQ * HDIM;
    const __bf16* vp = vt + (size_t)bh * HDIM * SEQ;

    int qrow0 = blockIdx.x * 128 + w * 16;       // local query-row base

    bf16x16 aq[4];
#pragma unroll
    for (int t = 0; t < 4; ++t) aq[t] = load_a_frag(qp, HDIM, qrow0, 32 * t, lane);

    floatx8 o[8];
#pragma unroll
    for (int j = 0; j < 8; ++j) o[j] = fzero();
    float m_run[8], l_run[8];
#pragma unroll
    for (int v = 0; v < 8; ++v) { m_run[v] = -INFINITY; l_run[v] = 0.0f; }

    const float scale = 0.08838834764831845f;    // 128^-0.5
    int cn = lane & 15, half = lane >> 4;

    for (int kb = 0; kb < SEQ; kb += 32) {
        // ---- S = Q K^T for 32 keys (two 16-key C tiles) --------------------
        floatx8 s0 = fzero(), s1 = fzero();
#pragma unroll
        for (int t = 0; t < 4; ++t) {
            bf16x16 bk0 = load_b_frag(kp, HDIM, kb,      32 * t, lane);
            bf16x16 bk1 = load_b_frag(kp, HDIM, kb + 16, 32 * t, lane);
            s0 = wmma_bf16(aq[t], bk0, s0);
            s1 = wmma_bf16(aq[t], bk1, s1);
        }

        // ---- online softmax (rows live across 16 lanes of a half-wave) ----
        float fac[8];
#pragma unroll
        for (int v = 0; v < 8; ++v) {
            float a = s0[v] * scale, b = s1[v] * scale;
            float mx = fmaxf(a, b);
            mx = fmaxf(mx, __shfl_xor(mx, 1, 32));
            mx = fmaxf(mx, __shfl_xor(mx, 2, 32));
            mx = fmaxf(mx, __shfl_xor(mx, 4, 32));
            mx = fmaxf(mx, __shfl_xor(mx, 8, 32));
            float mnew = fmaxf(m_run[v], mx);
            fac[v] = __expf(m_run[v] - mnew);
            m_run[v] = mnew;
            float p0 = __expf(a - mnew), p1 = __expf(b - mnew);
            float rs = p0 + p1;
            rs += __shfl_xor(rs, 1, 32);
            rs += __shfl_xor(rs, 2, 32);
            rs += __shfl_xor(rs, 4, 32);
            rs += __shfl_xor(rs, 8, 32);
            l_run[v] = l_run[v] * fac[v] + rs;
            s0[v] = p0; s1[v] = p1;
        }

        // ---- transpose P (C layout -> A layout) through LDS ---------------
#pragma unroll
        for (int v = 0; v < 8; ++v) {
            int m = v + 8 * half;
            Plds[w][m][cn]      = (__bf16)s0[v];
            Plds[w][m][16 + cn] = (__bf16)s1[v];
        }
        asm volatile("s_wait_dscnt 0" ::: "memory");
        bf16x16 ap = load_a_frag(&Plds[w][0][0], 32, 0, 0, lane);

        // ---- rescale and accumulate O += P * V ----------------------------
#pragma unroll
        for (int j = 0; j < 8; ++j) {
#pragma unroll
            for (int v = 0; v < 8; ++v) o[j][v] *= fac[v];
        }
#pragma unroll
        for (int j = 0; j < 8; ++j) {
            bf16x16 bv = load_b_frag(vp, SEQ, 16 * j, kb, lane);   // Vt contiguous
            o[j] = wmma_bf16(ap, bv, o[j]);
        }
    }

    // ---- normalize and write (b, n, c) fp32 -------------------------------
    float inv[8];
#pragma unroll
    for (int v = 0; v < 8; ++v) inv[v] = 1.0f / l_run[v];
#pragma unroll
    for (int j = 0; j < 8; ++j) {
        int c = hh * HDIM + j * 16 + cn;
#pragma unroll
        for (int v = 0; v < 8; ++v) {
            int nq = qrow0 + v + 8 * half;
            attn[((size_t)bidx * SEQ + nq) * DIM + c] = o[j][v] * inv[v];
        }
    }
}

// ---------------------------------------------------------------------------
// FFN GEMM with bias (+ optional exact GELU, bf16 out) or bias-only f32 out.
// Weight tiles staged in LDS by the TDM, double-buffered.
// ---------------------------------------------------------------------------
__global__ __launch_bounds__(256) void gemm_bias(const __bf16* __restrict__ A, int K,
                                                 const __bf16* __restrict__ W,
                                                 const float* __restrict__ bias,
                                                 __bf16* __restrict__ out_b,
                                                 float* __restrict__ out_f,
                                                 int N, int act) {
    __shared__ __align__(64) __bf16 Blds[2][64][32];   // 8 KB double buffer

    int lane = threadIdx.x & 31, w = threadIdx.x >> 5;
    int wm = w >> 1, wn = w & 1;
    int m0 = blockIdx.x * 128 + wm * 32;
    int nblk = blockIdx.y * 64;
    int n0 = nblk + wn * 32;

    floatx8 acc[2][2];
#pragma unroll
    for (int i = 0; i < 2; ++i)
#pragma unroll
        for (int j = 0; j < 2; ++j) acc[i][j] = fzero();

    const int ksteps = K / 32;
    if (w == 0) tdm_load_b_tile(W, K, N, nblk, 0, &Blds[0][0][0]);

    auto compute = [&](int i) {
        int k0 = i * 32;
        const __bf16* bbase = &Blds[i & 1][0][0];
        bf16x16 a0 = load_a_frag(A, K, m0, k0, lane);
        bf16x16 a1 = load_a_frag(A, K, m0 + 16, k0, lane);
        bf16x16 b0 = load_b_frag_lds(bbase, wn * 32, lane);
        bf16x16 b1 = load_b_frag_lds(bbase, wn * 32 + 16, lane);
        acc[0][0] = wmma_bf16(a0, b0, acc[0][0]);
        acc[0][1] = wmma_bf16(a0, b1, acc[0][1]);
        acc[1][0] = wmma_bf16(a1, b0, acc[1][0]);
        acc[1][1] = wmma_bf16(a1, b1, acc[1][1]);
    };

    for (int i = 0; i < ksteps - 1; ++i) {
        if (w == 0) {
            tdm_load_b_tile(W, K, N, nblk, (i + 1) * 32, &Blds[(i + 1) & 1][0][0]);
            __builtin_amdgcn_s_wait_tensorcnt(1);
        }
        __syncthreads();
        compute(i);
        __syncthreads();
    }
    if (w == 0) __builtin_amdgcn_s_wait_tensorcnt(0);
    __syncthreads();
    compute(ksteps - 1);

    int half = lane >> 4, cn = lane & 15;
#pragma unroll
    for (int i = 0; i < 2; ++i) {
#pragma unroll
        for (int j = 0; j < 2; ++j) {
            int col = n0 + j * 16 + cn;
            float bc = bias[col];
#pragma unroll
            for (int v = 0; v < 8; ++v) {
                int row = m0 + i * 16 + v + 8 * half;
                float t = acc[i][j][v] + bc;
                if (act) {
                    t = 0.5f * t * (1.0f + erff(t * 0.7071067811865476f));
                    out_b[(size_t)row * N + col] = (__bf16)t;
                } else {
                    out_f[(size_t)row * N + col] = t;
                }
            }
        }
    }
}

// ---------------------------------------------------------------------------
// Fused residual + LayerNorm: y = LN(a + b); writes f32 and optional bf16.
// One block per token row (896 elems, 256 threads).
// ---------------------------------------------------------------------------
__global__ __launch_bounds__(256) void ln_fused(const float* __restrict__ a,
                                                const float* __restrict__ b,
                                                const float* __restrict__ g,
                                                const float* __restrict__ be,
                                                float* __restrict__ of,
                                                __bf16* __restrict__ ob) {
    __shared__ float s1[256], s2[256];
    size_t row = blockIdx.x;
    const float* pa = a + row * DIM;
    const float* pb = b + row * DIM;
    int tid = threadIdx.x;

    float vals[4];
    float sum = 0.0f, sq = 0.0f;
#pragma unroll
    for (int i = 0; i < 4; ++i) {
        int idx = tid + i * 256;
        float v = 0.0f;
        if (idx < DIM) v = pa[idx] + pb[idx];
        vals[i] = v;
        sum += v; sq += v * v;
    }
    s1[tid] = sum; s2[tid] = sq;
    __syncthreads();
    for (int off = 128; off > 0; off >>= 1) {
        if (tid < off) { s1[tid] += s1[tid + off]; s2[tid] += s2[tid + off]; }
        __syncthreads();
    }
    float mu  = s1[0] * (1.0f / DIM);
    float var = s2[0] * (1.0f / DIM) - mu * mu;
    float rstd = rsqrtf(var + 1e-6f);
#pragma unroll
    for (int i = 0; i < 4; ++i) {
        int idx = tid + i * 256;
        if (idx < DIM) {
            float y = (vals[i] - mu) * rstd * g[idx] + be[idx];
            if (of) of[row * DIM + idx] = y;
            if (ob) ob[row * DIM + idx] = (__bf16)y;
        }
    }
}

// ---------------------------------------------------------------------------
// Host-side orchestration
// ---------------------------------------------------------------------------
extern "C" void kernel_launch(void* const* d_in, const int* in_sizes, int n_in,
                              void* d_out, int out_size, void* d_ws, size_t ws_size,
                              hipStream_t stream) {
    const float* x       = (const float*)d_in[0];
    const float* qkv_w   = (const float*)d_in[1];
    const float* proj1_w = (const float*)d_in[2];
    const float* proj1_b = (const float*)d_in[3];
    const float* proj2_w = (const float*)d_in[4];
    const float* proj2_b = (const float*)d_in[5];
    const float* ln1_g   = (const float*)d_in[6];
    const float* ln1_b   = (const float*)d_in[7];
    const float* ln2_g   = (const float*)d_in[8];
    const float* ln2_b   = (const float*)d_in[9];
    float* out = (float*)d_out;

    char* ws = (char*)d_ws;
    constexpr size_t N_XB   = (size_t)MTOK * DIM;       // 14680064
    constexpr size_t N_WQKV = (size_t)C3 * DIM;         // 2408448
    constexpr size_t N_W1   = (size_t)FFN * DIM;        // 1605632
    constexpr size_t N_W2   = (size_t)DIM * FFN;        // 1605632
    constexpr size_t N_QKV1 = (size_t)BATCH * HEADS * SEQ * HDIM; // 14680064
    constexpr size_t N_H1   = (size_t)MTOK * FFN;       // 29360128

    size_t off = 0;
    __bf16* xb   = (__bf16*)(ws + off); off += 2 * N_XB;
    __bf16* wqkv = (__bf16*)(ws + off); off += 2 * N_WQKV;
    __bf16* w1b  = (__bf16*)(ws + off); off += 2 * N_W1;
    __bf16* w2b  = (__bf16*)(ws + off); off += 2 * N_W2;
    __bf16* qb   = (__bf16*)(ws + off); off += 2 * N_QKV1;
    __bf16* kb   = (__bf16*)(ws + off); off += 2 * N_QKV1;
    __bf16* vtb  = (__bf16*)(ws + off); off += 2 * N_QKV1;
    float*  attn = (float*)(ws + off);  off += 4 * N_XB;  // reused as FFN out
    float*  x1   = (float*)(ws + off);  off += 4 * N_XB;
    __bf16* x1b  = (__bf16*)(ws + off); off += 2 * N_XB;
    __bf16* h1b  = (__bf16*)(ws + off); off += 2 * N_H1;
    (void)off; (void)ws_size; (void)in_sizes; (void)n_in; (void)out_size;

    // 1) casts to bf16
    cast_f32_bf16<<<(unsigned)(N_XB / 256),   256, 0, stream>>>(x, xb, (int)N_XB);
    cast_f32_bf16<<<(unsigned)(N_WQKV / 256), 256, 0, stream>>>(qkv_w, wqkv, (int)N_WQKV);
    cast_f32_bf16<<<(unsigned)(N_W1 / 256),   256, 0, stream>>>(proj1_w, w1b, (int)N_W1);
    cast_f32_bf16<<<(unsigned)(N_W2 / 256),   256, 0, stream>>>(proj2_w, w2b, (int)N_W2);

    // 2) QKV projection (scatter q/k row-major, v transposed)
    gemm_qkv<<<dim3(MTOK / 128, C3 / 64), 256, 0, stream>>>(xb, wqkv, qb, kb, vtb);

    // 3) flash attention -> attn (b, n, c) fp32
    attn_flash<<<dim3(SEQ / 128, BATCH * HEADS), 256, 0, stream>>>(qb, kb, vtb, attn);

    // 4) x1 = LN1(attn + x), plus bf16 copy for FFN
    ln_fused<<<MTOK, 256, 0, stream>>>(attn, x, ln1_g, ln1_b, x1, x1b);

    // 5) FFN up: gelu(x1 @ W1^T + b1) -> bf16
    gemm_bias<<<dim3(MTOK / 128, FFN / 64), 256, 0, stream>>>(
        x1b, DIM, w1b, proj1_b, h1b, nullptr, FFN, 1);

    // 6) FFN down: h1 @ W2^T + b2 -> fp32 (reuse attn buffer)
    gemm_bias<<<dim3(MTOK / 128, DIM / 64), 256, 0, stream>>>(
        h1b, FFN, w2b, proj2_b, nullptr, attn, DIM, 0);

    // 7) out = LN2(x1 + ffn)
    ln_fused<<<MTOK, 256, 0, stream>>>(attn, x1, ln2_g, ln2_b, out, nullptr);
}
